// distMPNN_layer_50027779064045
// MI455X (gfx1250) — matrix operand order repaired and checked by reading, not compiled, not run on hardware
//
#include <hip/hip_runtime.h>

// ---- MI455X / gfx1250: wave32, WMMA bf16 16x16x32 f32-acc, LDS-resident weights ----

typedef __attribute__((ext_vector_type(16))) __bf16       v16bf;
typedef __attribute__((ext_vector_type(8)))  float        v8f;
typedef __attribute__((ext_vector_type(4)))  unsigned int u32x4;
typedef __attribute__((ext_vector_type(4)))  float        f32x4;

#define IN_F  64
#define EF    32
#define OUT_F 32
#define WPB   10     // waves per block in edge kernel
#define EBLOCKS 512  // edge kernel max grid

union BF16x16 { u32x4 u[2]; v16bf v; __bf16 h[16]; };
union BF16x8  { u32x4 u;    __bf16 h[8]; };

__device__ __forceinline__ v8f wmma_bf16(v16bf a, v16bf b, v8f c) {
  return __builtin_amdgcn_wmma_f32_16x16x32_bf16(false, a, false, b, (short)0, c,
                                                 false, false);
}

// B fragment: 512 bf16, two 512-byte halves; lane stride 16 B (2-way max banks).
__device__ __forceinline__ v16bf load_frag_b(const __bf16* fb, int lane) {
  BF16x16 t;
  t.u[0] = *(const u32x4*)(fb + lane * 8);
  t.u[1] = *(const u32x4*)(fb + 256 + lane * 8);
  return t.v;
}

// A fragment from bf16 tile (row-major, ldc % 8 == 0):
// lane: row m = lane&15, K runs [k0..k0+7], [k0+16..k0+23], k0 = 8*(lane>>4).
__device__ __forceinline__ v16bf load_frag_a_lds(const __bf16* base, int ldc,
                                                 int coff, int lane) {
  int m = lane & 15, k0 = (lane >> 4) * 8;
  const __bf16* r = base + m * ldc + coff + k0;
  BF16x16 t;
  t.u[0] = *(const u32x4*)(r);
  t.u[1] = *(const u32x4*)(r + 16);
  return t.v;
}

// A fragment from global f32 row-major matrix with row clamp (converts to bf16).
__device__ __forceinline__ v16bf frag_a_global_rows(const float* mat, long row0,
                                                    long nrows, int ldc, int coff,
                                                    int lane) {
  int m = lane & 15, k0 = (lane >> 4) * 8;
  long row = row0 + m; if (row >= nrows) row = 0;
  const float* r = mat + row * (long)ldc + coff + k0;
  BF16x16 t;
#pragma unroll
  for (int i = 0; i < 8; ++i) { t.h[i] = (__bf16)r[i]; t.h[8 + i] = (__bf16)r[16 + i]; }
  return t.v;
}

__device__ __forceinline__ void cvt8_store(__bf16* dst, const float* s) {
  BF16x8 t;
#pragma unroll
  for (int i = 0; i < 8; ++i) t.h[i] = (__bf16)s[i];
  *(u32x4*)dst = t.u;
}

__device__ __forceinline__ void wave_fence() { asm volatile("" ::: "memory"); }

// ---- prepack W[outF][inF] -> B fragments of W^T (B[k][n] = W[n][k]) ----
__global__ void prepack_kernel(const float* __restrict__ W, __bf16* __restrict__ dst,
                               int outF, int inF, int ntMajor) {
  int idx = blockIdx.x * blockDim.x + threadIdx.x;
  int total = outF * inF;
  if (idx >= total) return;
  int frag = idx >> 9, within = idx & 511;
  int lane = within >> 4, nl = within & 15;
  int KC = inF >> 5, NT = outF >> 4;
  int kc, nt;
  if (ntMajor) { nt = frag / KC; kc = frag - nt * KC; }
  else         { kc = frag / NT; nt = frag - kc * NT; }
  int k = kc * 32 + (lane >> 4) * 8 + (nl & 7) + (nl >> 3) * 16, n = nt * 16 + (lane & 15);
  int delem = (nl < 8) ? (lane * 8 + nl) : (256 + lane * 8 + (nl - 8));
  dst[frag * 512 + delem] = (__bf16)W[(long)n * inF + k];
}

__global__ void zero_kernel(float* __restrict__ p, int n) {
  int i = blockIdx.x * blockDim.x + threadIdx.x;
  if (i < n) p[i] = 0.f;
}

// ---- h0 = node_feats @ W_node^T + b_node (16 nodes / wave) ----
__global__ __launch_bounds__(32) void node_embed_kernel(
    const float* __restrict__ nf, const __bf16* __restrict__ fragWn,
    const float* __restrict__ b_node, float* __restrict__ h0, int N) {
  int lane = threadIdx.x;
  long row0 = (long)blockIdx.x * 16;
  if (row0 >= N) return;
  v16bf a0 = frag_a_global_rows(nf, row0, N, IN_F, 0, lane);
  v16bf a1 = frag_a_global_rows(nf, row0, N, IN_F, 32, lane);
  int nl = lane & 15, hh = lane >> 4;
  bool full = (row0 + 16 <= (long)N);
#pragma unroll
  for (int nt = 0; nt < 2; ++nt) {
    v8f acc = {};
    acc = wmma_bf16(a0, load_frag_b(fragWn + (0 * 2 + nt) * 512, lane), acc);
    acc = wmma_bf16(a1, load_frag_b(fragWn + (1 * 2 + nt) * 512, lane), acc);
    float bias = b_node[nt * 16 + nl];
    if (full) {
#pragma unroll
      for (int v = 0; v < 8; ++v)
        h0[(row0 + v + 8 * hh) * OUT_F + nt * 16 + nl] = acc[v] + bias;
    } else {
#pragma unroll
      for (int v = 0; v < 8; ++v) {
        long row = row0 + v + 8 * hh;
        if (row < N) h0[row * OUT_F + nt * 16 + nl] = acc[v] + bias;
      }
    }
  }
}

// ---- per-edge fused f/m1/scale/m2 + scatter; weights LDS-resident ----
__global__ __launch_bounds__(WPB * 32) void edge_kernel(
    const float* __restrict__ h0, const float* __restrict__ ef,
    const int* __restrict__ src, const int* __restrict__ dstI,
    const __bf16* __restrict__ fragWe, const float* __restrict__ b_edge,
    const __bf16* __restrict__ fragM1, const float* __restrict__ b_m1,
    const __bf16* __restrict__ fragM2, const float* __restrict__ b_m2,
    float* __restrict__ agg, float* __restrict__ cnt, int E) {
  // 262144 B weights + 10 waves * 5120 B staging = 313344 B <= 320 KB
  __shared__ __align__(16) __bf16 wlds[131072];          // fragM1 (96K) + fragM2 (32K)
  __shared__ __align__(16) __bf16 msgS[WPB][16 * 104];   // msg_in; later reused as stage
  __shared__ __align__(16) __bf16 etileS[WPB][16 * 56];  // edge feats bf16

  // cooperative async fill: fragM1|fragM2 are contiguous (256 KB) in workspace
  {
    const char* g = (const char*)fragM1;
    unsigned lbase = (unsigned)(unsigned long long)(char*)wlds;  // LDS byte offset
    for (int i = threadIdx.x * 16; i < 262144; i += WPB * 32 * 16) {
      unsigned loff = lbase + (unsigned)i;
      const char* gp = g + i;
      asm volatile("global_load_async_to_lds_b128 %0, %1, off"
                   :: "v"(loff), "v"(gp) : "memory");
    }
    asm volatile("s_wait_asynccnt 0x0" ::: "memory");
  }
  __syncthreads();

  const __bf16* fM1 = wlds;              // 64 nt * 3 kc frags (nt-major)
  const __bf16* fM2 = wlds + 192 * 512;  // 32 kc * 2 nt frags (kc-major)

  int wave = threadIdx.x >> 5, lane = threadIdx.x & 31;
  int nl = lane & 15, hh = lane >> 4;
  __bf16* msg = msgS[wave];
  __bf16* stg = msg;  // stage overlays msg after msg A-frags are in VGPRs
  __bf16* etl = etileS[wave];

  long ntiles = ((long)E + 15) >> 4;
  for (long tile = (long)blockIdx.x * WPB + wave; tile < ntiles;
       tile += (long)gridDim.x * WPB) {
    int e0 = (int)(tile << 4);
    bool full = (e0 + 16 <= E);

    // gather h0[src] (half 0) / h0[dst] (half 1) rows -> msg cols [0..63]
    {
      int e = e0 + nl; if (e >= E) e = E - 1;
      int node = (hh == 0) ? src[e] : dstI[e];
      const float* hr = h0 + (long)node * OUT_F;
      float tmp[8];
#pragma unroll
      for (int c = 0; c < 32; c += 8) {
        f32x4 x = *(const f32x4*)(hr + c);
        f32x4 y = *(const f32x4*)(hr + c + 4);
        tmp[0] = x.x; tmp[1] = x.y; tmp[2] = x.z; tmp[3] = x.w;
        tmp[4] = y.x; tmp[5] = y.y; tmp[6] = y.z; tmp[7] = y.w;
        cvt8_store(&msg[nl * 104 + hh * 32 + c], tmp);
      }
    }
    // raw edge feature tile -> etl (bf16)
    {
      int e = e0 + nl; if (e >= E) e = E - 1;
      const float* er = ef + (long)e * EF;
      float tmp[8];
#pragma unroll
      for (int c = 0; c < 16; c += 8) {
        f32x4 x = *(const f32x4*)(er + hh * 16 + c);
        f32x4 y = *(const f32x4*)(er + hh * 16 + c + 4);
        tmp[0] = x.x; tmp[1] = x.y; tmp[2] = x.z; tmp[3] = x.w;
        tmp[4] = y.x; tmp[5] = y.y; tmp[6] = y.z; tmp[7] = y.w;
        cvt8_store(&etl[nl * 56 + hh * 16 + c], tmp);
      }
    }
    wave_fence();  // same-wave LDS ops are in-order; compiler fence only

    // f = edge @ W_edge^T + b_edge -> msg cols [64..95]
    {
      v16bf aE = load_frag_a_lds(etl, 56, 0, lane);
#pragma unroll
      for (int nt = 0; nt < 2; ++nt) {
        v8f acc = {};
        acc = wmma_bf16(aE, load_frag_b(fragWe + nt * 512, lane), acc);
        float bias = b_edge[nt * 16 + nl];
#pragma unroll
        for (int v = 0; v < 8; ++v)
          msg[(v + 8 * hh) * 104 + 64 + nt * 16 + nl] = (__bf16)(acc[v] + bias);
      }
    }
    wave_fence();

    v16bf aM0 = load_frag_a_lds(msg, 104, 0, lane);
    v16bf aM1 = load_frag_a_lds(msg, 104, 32, lane);
    v16bf aM2 = load_frag_a_lds(msg, 104, 64, lane);

    // d multipliers: d[e][j] = edge[e][j%32]; D-layout M = v+8*hh, col = p*16+nl
    float dv[2][8];
#pragma unroll
    for (int p = 0; p < 2; ++p)
#pragma unroll
      for (int v = 0; v < 8; ++v)
        dv[p][v] = (float)etl[(v + 8 * hh) * 56 + p * 16 + nl];
    wave_fence();  // aM/dv reads ordered before stage overwrites msg

    // main loop: two m2 k-chunks per iteration; stage = [16][64+8] inside msg
    v8f acc00 = {}, acc01 = {}, acc10 = {}, acc11 = {};
    for (int j2 = 0; j2 < 16; ++j2) {
#pragma unroll
      for (int q = 0; q < 4; ++q) {           // m1 n-tiles 4*j2 .. 4*j2+3
        int nt = 4 * j2 + q;
        const __bf16* fb = fM1 + (long)nt * 1536;
        v8f acc = {};
        acc = wmma_bf16(aM0, load_frag_b(fb,        lane), acc);
        acc = wmma_bf16(aM1, load_frag_b(fb + 512,  lane), acc);
        acc = wmma_bf16(aM2, load_frag_b(fb + 1024, lane), acc);
        float bias = b_m1[nt * 16 + nl];
#pragma unroll
        for (int v = 0; v < 8; ++v)
          stg[(v + 8 * hh) * 72 + q * 16 + nl] =
              (__bf16)((acc[v] + bias) * dv[q & 1][v]);
      }
      wave_fence();
      v16bf aSA = load_frag_a_lds(stg, 72, 0, lane);   // k-chunk 2*j2
      v16bf aSB = load_frag_a_lds(stg, 72, 32, lane);  // k-chunk 2*j2+1
      const __bf16* f2 = fM2 + (long)(4 * j2) * 512;
      acc00 = wmma_bf16(aSA, load_frag_b(f2,        lane), acc00);
      acc01 = wmma_bf16(aSA, load_frag_b(f2 + 512,  lane), acc01);
      acc10 = wmma_bf16(aSB, load_frag_b(f2 + 1024, lane), acc10);
      acc11 = wmma_bf16(aSB, load_frag_b(f2 + 1536, lane), acc11);
      wave_fence();
    }
    v8f accO0 = acc00 + acc10;
    v8f accO1 = acc01 + acc11;

    // scatter segment-sum to dst nodes
    float b0 = b_m2[nl], b1 = b_m2[16 + nl];
    if (full) {
#pragma unroll
      for (int v = 0; v < 8; ++v) {
        int node = dstI[e0 + v + 8 * hh];
        atomicAdd(&agg[(long)node * OUT_F + nl],      accO0[v] + b0);
        atomicAdd(&agg[(long)node * OUT_F + 16 + nl], accO1[v] + b1);
      }
      if (lane < 16) atomicAdd(&cnt[dstI[e0 + lane]], 1.0f);
    } else {
#pragma unroll
      for (int v = 0; v < 8; ++v) {
        int e = e0 + v + 8 * hh;
        if (e < E) {
          int node = dstI[e];
          atomicAdd(&agg[(long)node * OUT_F + nl],      accO0[v] + b0);
          atomicAdd(&agg[(long)node * OUT_F + 16 + nl], accO1[v] + b1);
        }
      }
      if (lane < 16 && e0 + lane < E) atomicAdd(&cnt[dstI[e0 + lane]], 1.0f);
    }
  }
}

// ---- mean/relu/concat/m3 + residual (16 nodes / wave) ----
__global__ __launch_bounds__(32) void finalize_kernel(
    const float* __restrict__ h0, const float* __restrict__ agg,
    const float* __restrict__ cnt, const __bf16* __restrict__ fragW3,
    const float* __restrict__ b_m3, float* __restrict__ out, int N) {
  __shared__ __align__(16) __bf16 cat[16 * 72];  // [16][64 + pad]: h0 | relu(h)
  __shared__ __align__(16) float  h0t[16 * 36];
  int lane = threadIdx.x;
  long row0 = (long)blockIdx.x * 16;
  if (row0 >= N) return;
  int nl = lane & 15, hh = lane >> 4;
  {
    long node = row0 + nl; if (node >= N) node = N - 1;
    float c = cnt[node];
    float inv = (c > 0.f) ? (1.f / c) : 0.f;
    const float* ar = agg + node * OUT_F + hh * 16;
    const float* hr = h0  + node * OUT_F + hh * 16;
#pragma unroll
    for (int c4 = 0; c4 < 16; c4 += 4) {
      f32x4 av = *(const f32x4*)(ar + c4);
      f32x4 hv = *(const f32x4*)(hr + c4);
      float ax[4] = {av.x, av.y, av.z, av.w};
      float hx[4] = {hv.x, hv.y, hv.z, hv.w};
#pragma unroll
      for (int i = 0; i < 4; ++i) {
        float m = (c > 0.f) ? ax[i] * inv : hx[i];
        m = fmaxf(m, 0.f);
        cat[nl * 72 + hh * 16 + c4 + i]      = (__bf16)hx[i];
        cat[nl * 72 + 32 + hh * 16 + c4 + i] = (__bf16)m;
        h0t[nl * 36 + hh * 16 + c4 + i]      = hx[i];
      }
    }
  }
  __syncthreads();
  v16bf a0 = load_frag_a_lds(cat, 72, 0, lane);
  v16bf a1 = load_frag_a_lds(cat, 72, 32, lane);
  bool full = (row0 + 16 <= (long)N);
#pragma unroll
  for (int nt = 0; nt < 2; ++nt) {
    v8f acc = {};
    acc = wmma_bf16(a0, load_frag_b(fragW3 + (0 * 2 + nt) * 512, lane), acc);
    acc = wmma_bf16(a1, load_frag_b(fragW3 + (1 * 2 + nt) * 512, lane), acc);
    float bias = b_m3[nt * 16 + nl];
    if (full) {
#pragma unroll
      for (int v = 0; v < 8; ++v)
        out[(row0 + v + 8 * hh) * OUT_F + nt * 16 + nl] =
            h0t[(v + 8 * hh) * 36 + nt * 16 + nl] + acc[v] + bias;
    } else {
#pragma unroll
      for (int v = 0; v < 8; ++v) {
        long row = row0 + v + 8 * hh;
        if (row < N)
          out[row * OUT_F + nt * 16 + nl] =
              h0t[(v + 8 * hh) * 36 + nt * 16 + nl] + acc[v] + bias;
      }
    }
  }
}

extern "C" void kernel_launch(void* const* d_in, const int* in_sizes, int n_in,
                              void* d_out, int out_size, void* d_ws, size_t ws_size,
                              hipStream_t stream) {
  (void)n_in; (void)out_size; (void)ws_size;
  const float* node_feats = (const float*)d_in[0];
  const float* edge_feats = (const float*)d_in[1];
  const int*   src        = (const int*)d_in[2];
  const int*   dstI       = (const int*)d_in[3];
  const float* W_node = (const float*)d_in[4];
  const float* b_node = (const float*)d_in[5];
  const float* W_edge = (const float*)d_in[6];
  const float* b_edge = (const float*)d_in[7];
  const float* W_m1   = (const float*)d_in[8];
  const float* b_m1   = (const float*)d_in[9];
  const float* W_m2   = (const float*)d_in[10];
  const float* b_m2   = (const float*)d_in[11];
  const float* W_m3   = (const float*)d_in[12];
  const float* b_m3   = (const float*)d_in[13];

  int N = in_sizes[0] / IN_F;   // 100000
  int E = in_sizes[1] / EF;     // 400000

  // workspace layout (~26.3 MB; all bases 16-byte aligned)
  float*  h0     = (float*)d_ws;
  float*  agg    = h0 + (long)N * OUT_F;
  float*  cnt    = agg + (long)N * OUT_F;
  __bf16* fragWn = (__bf16*)(cnt + N);
  __bf16* fragWe = fragWn + 4 * 512;     // W_node: 2kc x 2nt
  __bf16* fragM1 = fragWe + 2 * 512;     // W_edge: 1kc x 2nt
  __bf16* fragM2 = fragM1 + 192 * 512;   // W_m1 : 64nt x 3kc (nt-major); contiguous w/ fragM2
  __bf16* fragW3 = fragM2 + 64 * 512;    // W_m2 : 32kc x 2nt

  prepack_kernel<<<(32 * 64   + 255) / 256, 256, 0, stream>>>(W_node, fragWn, 32, 64, 0);
  prepack_kernel<<<(32 * 32   + 255) / 256, 256, 0, stream>>>(W_edge, fragWe, 32, 32, 0);
  prepack_kernel<<<(1024 * 96 + 255) / 256, 256, 0, stream>>>(W_m1,   fragM1, 1024, 96, 1);
  prepack_kernel<<<(32 * 1024 + 255) / 256, 256, 0, stream>>>(W_m2,   fragM2, 32, 1024, 0);
  prepack_kernel<<<(32 * 64   + 255) / 256, 256, 0, stream>>>(W_m3,   fragW3, 32, 64, 0);

  int zn = N * (OUT_F + 1);  // agg and cnt contiguous
  zero_kernel<<<(zn + 255) / 256, 256, 0, stream>>>(agg, zn);

  node_embed_kernel<<<(N + 15) / 16, 32, 0, stream>>>(node_feats, fragWn, b_node, h0, N);

  long ntiles = ((long)E + 15) / 16;
  int eblocks = (int)((ntiles + WPB - 1) / WPB);
  if (eblocks > EBLOCKS) eblocks = EBLOCKS;
  edge_kernel<<<eblocks, WPB * 32, 0, stream>>>(h0, edge_feats, src, dstI,
                                                fragWe, b_edge, fragM1, b_m1,
                                                fragM2, b_m2, agg, cnt, E);

  finalize_kernel<<<(N + 15) / 16, 32, 0, stream>>>(h0, agg, cnt, fragW3, b_m3,
                                                    (float*)d_out, N);
}
